// MultiHeadAttention_446676599161
// MI455X (gfx1250) — compile-verified, matrix operand order checked by reading, not run on hardware
//
#include <hip/hip_runtime.h>
#include <hip/hip_bf16.h>

#define B_  4
#define S_  2048
#define DM  1024
#define H_  16
#define DK  64
#define DV  64
#define SPAD 2052   // LDS score-row stride (floats): pad so C-frag stores are bank-conflict-free

typedef __bf16 bf16;
typedef __attribute__((ext_vector_type(16))) __bf16 v16bf;
typedef __attribute__((ext_vector_type(8)))  __bf16 v8bf;
typedef __attribute__((ext_vector_type(8)))  float  v8f;
typedef __attribute__((ext_vector_type(4)))  float  v4f;

__device__ __forceinline__ v16bf frag_combine(v8bf lo, v8bf hi) {
  return __builtin_shufflevector(lo, hi, 0,1,2,3,4,5,6,7,8,9,10,11,12,13,14,15);
}

// 16x32 bf16 operand fragment per ISA A-layout:
//   lane<16 : row=lane,    elems = K[0..7]  then K[16..23]
//   lane>=16: row=lane-16, elems = K[8..15] then K[24..31]
// (B operand assumed symmetric with rows == N.)
__device__ __forceinline__ v16bf load_frag_bf16(const bf16* p, int stride) {
  int lane = threadIdx.x & 31;
  int r    = lane & 15;
  int koff = (lane < 16) ? 0 : 8;
  const bf16* q = p + (size_t)r * stride + koff;
  v8bf lo = *(const v8bf*)(q);
  v8bf hi = *(const v8bf*)(q + 16);
  return frag_combine(lo, hi);
}

// Same fragment from an fp32 source (converted to bf16 in-register).
__device__ __forceinline__ v16bf load_frag_f32(const float* p, int stride) {
  int lane = threadIdx.x & 31;
  int r    = lane & 15;
  int koff = (lane < 16) ? 0 : 8;
  const float* q = p + (size_t)r * stride + koff;
  v4f x0 = *(const v4f*)(q);
  v4f x1 = *(const v4f*)(q + 4);
  v4f x2 = *(const v4f*)(q + 16);
  v4f x3 = *(const v4f*)(q + 20);
  v8bf lo, hi;
  lo[0]=(bf16)x0.x; lo[1]=(bf16)x0.y; lo[2]=(bf16)x0.z; lo[3]=(bf16)x0.w;
  lo[4]=(bf16)x1.x; lo[5]=(bf16)x1.y; lo[6]=(bf16)x1.z; lo[7]=(bf16)x1.w;
  hi[0]=(bf16)x2.x; hi[1]=(bf16)x2.y; hi[2]=(bf16)x2.z; hi[3]=(bf16)x2.w;
  hi[4]=(bf16)x3.x; hi[5]=(bf16)x3.y; hi[6]=(bf16)x3.z; hi[7]=(bf16)x3.w;
  return frag_combine(lo, hi);
}

__device__ __forceinline__ v8f wmma_bf16(v16bf a, v16bf b, v8f c) {
  return __builtin_amdgcn_wmma_f32_16x16x32_bf16(
      /*neg_a=*/false, a, /*neg_b=*/false, b,
      /*c_mod=*/(short)0, c, /*reuse_a=*/false, /*reuse_b=*/false);
}

// CDNA5 async global->LDS copy of 16 bytes per lane (ASYNCcnt-tracked).
// ldsoff: 32-bit LDS byte address (addr[31:0] of the generic pointer).
__device__ __forceinline__ void async_copy_b128(unsigned ldsoff, const void* gptr) {
  asm volatile("global_load_async_to_lds_b128 %0, %1, off"
               :: "v"(ldsoff), "v"((unsigned long long)gptr) : "memory");
}
__device__ __forceinline__ void wait_async0() {
  asm volatile("s_wait_asynccnt 0" ::: "memory");
}
__device__ __forceinline__ unsigned lds_addr(const void* p) {
  return (unsigned)(size_t)p;   // shared aperture: addr[31:0] == LDS offset
}

// ---------------------------------------------------------------------------
// fp32 -> bf16 elementwise conversion (one up-front pass; keeps GEMM hot loop
// free of cvt VALU and makes tile staging a pure async copy).
// ---------------------------------------------------------------------------
__global__ __launch_bounds__(256)
void cvt_f32_bf16(const float* __restrict__ in, bf16* __restrict__ out) {
  const size_t i = ((size_t)blockIdx.x * 256 + threadIdx.x) * 8;
  v4f x0 = *(const v4f*)&in[i];
  v4f x1 = *(const v4f*)&in[i + 4];
  v8bf y;
  y[0]=(bf16)x0.x; y[1]=(bf16)x0.y; y[2]=(bf16)x0.z; y[3]=(bf16)x0.w;
  y[4]=(bf16)x1.x; y[5]=(bf16)x1.y; y[6]=(bf16)x1.z; y[7]=(bf16)x1.w;
  *(v8bf*)&out[i] = y;
}

// ---------------------------------------------------------------------------
// GEMM: Out = A(M x K) @ W^T, both bf16; W is (N x K) row-major.
// BM=128, BN=64, BK=64; double-buffered LDS staged with async-to-LDS copies;
// rows padded to 72 elems (144 B = 36 banks) so fragment ds_load_b128s are
// bank-conflict-free. 8 waves = 4M x 2N, 8 WMMAs per K-step per wave.
// OUT_MODE 0: bf16 row-major [M][N]          (Q/K projections)
// OUT_MODE 1: bf16 transposed [b][h][dv][s]  (V projection)
// OUT_MODE 2: fp32 row-major [M][N]          (FC projection -> d_out)
// ---------------------------------------------------------------------------
template <int OUT_MODE>
__global__ __launch_bounds__(256)
void gemm_bf16(const bf16* __restrict__ A, const bf16* __restrict__ Wb,
               void* __restrict__ Out) {
  constexpr int N = DM, K = DM;
  constexpr int BM = 128, BN = 64, BK = 64;
  constexpr int LDK = BK + 8;                 // padded row stride (bf16 elems)
  __shared__ bf16 As[2][BM * LDK];            // 2 x 18 KB
  __shared__ bf16 Bs[2][BN * LDK];            // 2 x  9 KB

  const int tid = threadIdx.x;
  const int n0  = blockIdx.x * BN;
  const int m0  = blockIdx.y * BM;
  const int wid = tid >> 5;
  const int wm  = (wid & 3) * 32;             // 4 waves over M
  const int wn  = (wid >> 2) * 32;            // 2 waves over N

  // Per-thread staging chunks (16 B each): A = 1024 chunks (4/thread),
  // B = 512 chunks (2/thread).
  const int arow = tid >> 3, acol = (tid & 7) * 8;

  auto stage = [&](int buf, int k0) {
    #pragma unroll
    for (int i = 0; i < 4; ++i) {
      int row = arow + i * 32;
      async_copy_b128(lds_addr(&As[buf][row * LDK + acol]),
                      &A[(size_t)(m0 + row) * K + k0 + acol]);
    }
    #pragma unroll
    for (int i = 0; i < 2; ++i) {
      int row = arow + i * 32;
      async_copy_b128(lds_addr(&Bs[buf][row * LDK + acol]),
                      &Wb[(size_t)(n0 + row) * K + k0 + acol]);
    }
  };

  v8f c00 = {0,0,0,0,0,0,0,0}, c01 = c00, c10 = c00, c11 = c00;

  stage(0, 0);
  wait_async0();
  __syncthreads();

  constexpr int NKT = K / BK;                 // 16 iterations
  for (int kt = 0; kt < NKT; ++kt) {
    const int cur = kt & 1;
    if (kt + 1 < NKT) stage(cur ^ 1, (kt + 1) * BK);
    #pragma unroll
    for (int ks = 0; ks < 2; ++ks) {
      const int ko = ks * 32;
      v16bf a0 = load_frag_bf16(&As[cur][(wm     ) * LDK + ko], LDK);
      v16bf a1 = load_frag_bf16(&As[cur][(wm + 16) * LDK + ko], LDK);
      v16bf b0 = load_frag_bf16(&Bs[cur][(wn     ) * LDK + ko], LDK);
      v16bf b1 = load_frag_bf16(&Bs[cur][(wn + 16) * LDK + ko], LDK);
      c00 = wmma_bf16(a0, b0, c00);
      c01 = wmma_bf16(a0, b1, c01);
      c10 = wmma_bf16(a1, b0, c10);
      c11 = wmma_bf16(a1, b1, c11);
    }
    wait_async0();          // next buffer fully in LDS (this wave's share)
    __syncthreads();        // everyone's share visible
  }

  // Epilogue. C layout: VGPR g -> (M = g + (lane<16?0:8), N = lane&15).
  const int lane  = tid & 31;
  const int nl    = lane & 15;
  const int mbase = (lane < 16) ? 0 : 8;
  #pragma unroll
  for (int f = 0; f < 4; ++f) {
    const v8f c = (f == 0) ? c00 : (f == 1) ? c01 : (f == 2) ? c10 : c11;
    const int fm = (f >> 1) * 16, fn = (f & 1) * 16;
    #pragma unroll
    for (int g = 0; g < 8; ++g) {
      int m = m0 + wm + fm + mbase + g;
      int n = n0 + wn + fn + nl;
      float v = c[g];
      if (OUT_MODE == 0) {
        ((bf16*)Out)[(size_t)m * N + n] = (bf16)v;
      } else if (OUT_MODE == 1) {
        int bb = m >> 11, s = m & (S_ - 1);
        int hh = n >> 6,  d = n & 63;
        ((bf16*)Out)[((((size_t)bb * H_ + hh) * DV + d) << 11) + s] = (bf16)v;
      } else {
        ((float*)Out)[(size_t)m * N + n] = v;
      }
    }
  }
}

// ---------------------------------------------------------------------------
// Fused attention: per (b, h, 16-query tile), with the 16x2048 score block
// resident in LDS. Phase 1: QK^T scores (WMMA) -> LDS. Phase 2: softmax in
// LDS + single coalesced fp32 write of attn probs to d_out. Phase 3:
// ctx = P @ V (WMMA, A-fragments pulled from LDS, f32->bf16 on the fly).
// attn tensor touches HBM exactly once (the mandated 1 GB output write).
// ---------------------------------------------------------------------------
__global__ __launch_bounds__(256)
void attn_fused(const bf16* __restrict__ qh, const bf16* __restrict__ kh,
                const bf16* __restrict__ vht, const int* __restrict__ mask,
                float* __restrict__ attn, bf16* __restrict__ ctx) {
  extern __shared__ char smem[];
  float* Ss = (float*)smem;                       // 16 x SPAD fp32 (128.3 KB)
  bf16*  Qs = (bf16*)(smem + 16 * SPAD * 4);      // 16 x 64 bf16 (2 KB)

  const int b = blockIdx.z, h = blockIdx.y, q0 = blockIdx.x * 16;
  const int tid = threadIdx.x, wid = tid >> 5, lane = tid & 31;
  const int nl = lane & 15, mbase = (lane < 16) ? 0 : 8;

  // Stage Q tile.
  {
    int row = tid >> 4, c = (tid & 15) * 4;
    *(uint2*)&Qs[row * 64 + c] =
        *(const uint2*)&qh[((size_t)(b * S_ + q0 + row)) * DM + h * DK + c];
  }
  __syncthreads();

  const v16bf a0 = load_frag_bf16(Qs, 64);        // dk 0..31
  const v16bf a1 = load_frag_bf16(Qs + 32, 64);   // dk 32..63

  // ---- Phase 1: scores -> LDS (each wave owns 256 keys) ----
  for (int t = 0; t < 16; ++t) {
    const int kbase = wid * 256 + t * 16;
    const bf16* kp = &kh[((size_t)(b * S_ + kbase)) * DM + h * DK];
    v16bf b0 = load_frag_bf16(kp,      DM);
    v16bf b1 = load_frag_bf16(kp + 32, DM);
    v8f c = {0,0,0,0,0,0,0,0};
    c = wmma_bf16(a0, b0, c);
    c = wmma_bf16(a1, b1, c);
    #pragma unroll
    for (int g = 0; g < 8; ++g) {
      int qq = mbase + g;
      int kk = kbase + nl;
      float sv = c[g] * 0.125f;                 // 1/sqrt(64)
      if (mask[b * S_ + kk]) sv = -1e6f;
      Ss[qq * SPAD + kk] = sv;
    }
  }
  __syncthreads();

  // ---- Phase 2: softmax (one wave per 2 rows), write probs to LDS+global ----
  const size_t arow = (((size_t)b * H_ + h) * S_ + q0) * (size_t)S_;
  #pragma unroll
  for (int rr = 0; rr < 2; ++rr) {
    const int r = wid * 2 + rr;
    float* row = Ss + r * SPAD;
    v4f x[16];
    float mx = -1e30f;
    #pragma unroll
    for (int j = 0; j < 16; ++j) {
      x[j] = *(const v4f*)&row[j * 128 + lane * 4];
      mx = fmaxf(mx, fmaxf(fmaxf(x[j].x, x[j].y), fmaxf(x[j].z, x[j].w)));
    }
    #pragma unroll
    for (int off = 16; off > 0; off >>= 1) mx = fmaxf(mx, __shfl_xor(mx, off, 32));
    float sum = 0.f;
    #pragma unroll
    for (int j = 0; j < 16; ++j) {
      x[j].x = __expf(x[j].x - mx); x[j].y = __expf(x[j].y - mx);
      x[j].z = __expf(x[j].z - mx); x[j].w = __expf(x[j].w - mx);
      sum += (x[j].x + x[j].y) + (x[j].z + x[j].w);
    }
    #pragma unroll
    for (int off = 16; off > 0; off >>= 1) sum += __shfl_xor(sum, off, 32);
    const float inv = 1.0f / sum;
    #pragma unroll
    for (int j = 0; j < 16; ++j) {
      v4f y;
      y.x = x[j].x * inv; y.y = x[j].y * inv;
      y.z = x[j].z * inv; y.w = x[j].w * inv;
      *(v4f*)&row[j * 128 + lane * 4] = y;
      *(v4f*)&attn[arow + (size_t)r * S_ + j * 128 + lane * 4] = y;
    }
  }
  __syncthreads();

  // ---- Phase 3: ctx = P @ V.  vht is [b][h][dv][s] so B rows == N. ----
  const int nt = wid & 3;        // dv tile: nt*16
  const int ks = wid >> 2;       // k slice: ks*1024
  const bf16* vp = vht + (((size_t)b * H_ + h) * DV + nt * 16) * (size_t)S_;
  v8f c = {0,0,0,0,0,0,0,0};
  for (int kk = 0; kk < 32; ++kk) {
    const int k0 = ks * 1024 + kk * 32;
    v16bf a  = load_frag_f32(Ss + k0, SPAD);     // LDS-resident probs
    v16bf bb = load_frag_bf16(vp + k0, S_);
    c = wmma_bf16(a, bb, c);
  }
  __syncthreads();               // done reading Ss; reuse it for reduction
  float* cred = Ss;              // 8 x 256 floats
  #pragma unroll
  for (int g = 0; g < 8; ++g) cred[wid * 256 + (mbase + g) * 16 + nl] = c[g];
  __syncthreads();
  if (ks == 0) {
    #pragma unroll
    for (int g = 0; g < 8; ++g) {
      const int ml = mbase + g;
      float v = c[g] + cred[(wid + 4) * 256 + ml * 16 + nl];
      ctx[((size_t)(b * S_ + q0 + ml)) * DM + h * DV + nt * 16 + nl] = (bf16)v;
    }
  }
}

// ---------------------------------------------------------------------------
// out = LayerNorm(out + residual) * gamma + beta. One wave per 1024-wide row.
// ---------------------------------------------------------------------------
__global__ __launch_bounds__(256)
void ln_residual(float* __restrict__ out, const float* __restrict__ resid,
                 const float* __restrict__ gamma, const float* __restrict__ beta) {
  const int wid = threadIdx.x >> 5, lane = threadIdx.x & 31;
  const int row = blockIdx.x * 8 + wid;
  float* o = out + (size_t)row * DM;
  const float* r = resid + (size_t)row * DM;
  float x[32], sum = 0.f, sq = 0.f;
  #pragma unroll
  for (int j = 0; j < 32; ++j) {
    float v = o[lane + j * 32] + r[lane + j * 32];
    x[j] = v; sum += v; sq += v * v;
  }
  #pragma unroll
  for (int off = 16; off > 0; off >>= 1) {
    sum += __shfl_xor(sum, off, 32);
    sq  += __shfl_xor(sq,  off, 32);
  }
  const float mu  = sum * (1.0f / DM);
  const float var = sq * (1.0f / DM) - mu * mu;
  const float rs  = rsqrtf(var + 1e-5f);
  #pragma unroll
  for (int j = 0; j < 32; ++j) {
    const int i = lane + j * 32;
    o[i] = (x[j] - mu) * rs * gamma[i] + beta[i];
  }
}

// ---------------------------------------------------------------------------
extern "C" void kernel_launch(void* const* d_in, const int* in_sizes, int n_in,
                              void* d_out, int out_size, void* d_ws, size_t ws_size,
                              hipStream_t stream) {
  const float* q    = (const float*)d_in[0];
  const float* k    = (const float*)d_in[1];
  const float* v    = (const float*)d_in[2];
  const int*   mask = (const int*)d_in[3];
  const float* w_q  = (const float*)d_in[4];
  const float* w_k  = (const float*)d_in[5];
  const float* w_v  = (const float*)d_in[6];
  const float* w_fc = (const float*)d_in[7];
  const float* ln_g = (const float*)d_in[8];
  const float* ln_b = (const float*)d_in[9];

  float* out  = (float*)d_out;
  float* attn = out + (size_t)B_ * S_ * DM;   // [B,H,S,S] fp32 (~1 GB)

  const size_t ACT = (size_t)B_ * S_ * DM;    // 8.4M elems
  const size_t WSZ = (size_t)DM * DM;         // 1M elems
  bf16* xq  = (bf16*)d_ws;                    // bf16 activations
  bf16* xk  = xq  + ACT;
  bf16* xv  = xk  + ACT;
  bf16* wqb = xv  + ACT;                      // bf16 weights
  bf16* wkb = wqb + WSZ;
  bf16* wvb = wkb + WSZ;
  bf16* wfb = wvb + WSZ;
  bf16* qh  = wfb + WSZ;                      // projected heads
  bf16* kh  = qh  + ACT;
  bf16* vht = kh  + ACT;                      // [b][h][dv][s]
  bf16* ctx = vht + ACT;                      // [b][s][h*dv]

  // One-time fp32 -> bf16 conversions.
  cvt_f32_bf16<<<dim3(ACT / 2048), 256, 0, stream>>>(q, xq);
  cvt_f32_bf16<<<dim3(ACT / 2048), 256, 0, stream>>>(k, xk);
  cvt_f32_bf16<<<dim3(ACT / 2048), 256, 0, stream>>>(v, xv);
  cvt_f32_bf16<<<dim3(WSZ / 2048), 256, 0, stream>>>(w_q,  wqb);
  cvt_f32_bf16<<<dim3(WSZ / 2048), 256, 0, stream>>>(w_k,  wkb);
  cvt_f32_bf16<<<dim3(WSZ / 2048), 256, 0, stream>>>(w_v,  wvb);
  cvt_f32_bf16<<<dim3(WSZ / 2048), 256, 0, stream>>>(w_fc, wfb);

  const dim3 gg(DM / 64, (B_ * S_) / 128);    // (N tiles, M tiles)
  gemm_bf16<0><<<gg, 256, 0, stream>>>(xq, wqb, qh);
  gemm_bf16<0><<<gg, 256, 0, stream>>>(xk, wkb, kh);
  gemm_bf16<1><<<gg, 256, 0, stream>>>(xv, wvb, vht);

  const size_t lds_bytes = (size_t)16 * SPAD * 4 + 16 * 64 * 2;  // 133,376 B
  attn_fused<<<dim3(S_ / 16, H_, B_), 256, lds_bytes, stream>>>(
      qh, kh, vht, mask, attn, ctx);

  gemm_bf16<2><<<gg, 256, 0, stream>>>(ctx, wfb, out);
  ln_residual<<<dim3(B_ * S_ / 8), 256, 0, stream>>>(out, q, ln_g, ln_b);
}
